// VectorQuantizer_11072425689459
// MI455X (gfx1250) — matrix operand order repaired and checked by reading, not compiled, not run on hardware
//
#include <hip/hip_runtime.h>
#include <hip/hip_bf16.h>

typedef __attribute__((ext_vector_type(16))) _Float16 v16h;
typedef __attribute__((ext_vector_type(8)))  float    v8f;

#define VQ_N        524288
#define VQ_D        64
#define VQ_K        512
#define VQ_NTILES   (VQ_N / 16)        // 32768 row tiles of 16
#define VQ_BLOCKS   1024
#define VQ_TPB      256                // 8 waves (wave32)
#define VQ_WAVES    (VQ_BLOCKS * (VQ_TPB / 32))
#define COMMIT      0.25f

// ---------------------------------------------------------------------------
// Kernel 0: per-code squared norms ||e_k||^2 -> d_ws[0..511]
// ---------------------------------------------------------------------------
__global__ __launch_bounds__(256) void vq_en2_kernel(const float* __restrict__ cb,
                                                     float* __restrict__ en2) {
    int c = blockIdx.x * blockDim.x + threadIdx.x;
    if (c < VQ_K) {
        const float* p = cb + (size_t)c * VQ_D;
        float s = 0.0f;
#pragma unroll
        for (int d = 0; d < VQ_D; ++d) s += p[d] * p[d];
        en2[c] = s;
    }
}

// ---------------------------------------------------------------------------
// Kernel 1: fused VQ kernel. Distances via v_wmma_f32_16x16x32_f16:
//   d = x_tile(16x64,f16) * (-2*E)^T(64x16,f16)  (C = inline 0)
//   dist = d + ||e||^2   (||x||^2 dropped; constant per row)
// Explicit 2-stage modulo schedule (d0/d1, rolled loop, unroll(1)): each
// argmin update is one full column-tile behind its producing WMMA, hiding the
// WMMA->VALU hazard window without register-pressure blowup.
// Cross-lane argmin: branchless reduce on sortable u64 keys
// (ordered-float << 32 | code) -> exact first-index tie-break.
// ---------------------------------------------------------------------------
__global__ __launch_bounds__(VQ_TPB) void vq_main_kernel(
    const float* __restrict__ x,
    const float* __restrict__ cb,
    const float* __restrict__ en2g,
    float* __restrict__ out,
    int*   __restrict__ idxOut,
    float* __restrict__ partials) {

    __shared__ _Float16 sB[VQ_K * VQ_D];   // -2*codebook in f16, 64 KB LDS

    const int tid  = threadIdx.x;
    const int lane = tid & 31;
    const int wave = tid >> 5;
    const int col  = lane & 15;            // code residue / row residue
    const int hi   = lane >> 4;            // half-wave select

    // Stage -2*E into LDS as f16 (coalesced f32 reads).
    for (int i = tid; i < VQ_K * VQ_D; i += VQ_TPB)
        sB[i] = (_Float16)(-2.0f * cb[i]);
    __syncthreads();

    // ||e||^2 for this lane's 32 candidate codes (code = t*16 + col).
    float en2reg[32];
#pragma unroll
    for (int t = 0; t < 32; ++t) en2reg[t] = en2g[t * 16 + col];

    const int gwave  = blockIdx.x * (VQ_TPB / 32) + wave;
    const int nwaves = gridDim.x * (VQ_TPB / 32);

    float lossAcc = 0.0f;

    for (int tile = gwave; tile < VQ_NTILES; tile += nwaves) {
        const int r0 = tile * 16;
        const float* rowp = x + (size_t)(r0 + col) * VQ_D;

        if (tile + nwaves < VQ_NTILES)   // uniform branch -> global_prefetch_b8
            __builtin_prefetch(x + (size_t)(tile + nwaves) * 16 * VQ_D
                                 + (size_t)col * VQ_D, 0, 0);

        // A fragments (16x32 f16, contiguous k order, same as B):
        //  lane<16: row=lane, K={0..15}; lane>=16: row=lane-16, K={16..31}
        v16h a0, a1;
#pragma unroll
        for (int s = 0; s < 2; ++s) {
            const float* p0 = rowp + s * 32 + hi * 16;
            const float* p1 = rowp + s * 32 + hi * 16 + 8;
            v16h a;
#pragma unroll
            for (int j = 0; j < 8; ++j) a[j]     = (_Float16)p0[j];
#pragma unroll
            for (int j = 0; j < 8; ++j) a[8 + j] = (_Float16)p1[j];
            if (s == 0) a0 = a; else a1 = a;
        }

        float minv[8];
        int   mini[8];
#pragma unroll
        for (int r = 0; r < 8; ++r) { minv[r] = 3.4e38f; mini[r] = 0; }

        // Distance for one 16-code column tile (2 chained WMMAs, C = inline 0).
        auto dist_tile = [&](int t) -> v8f {
            const int cc = t * 16 + col;
            const v16h b0 = *(const v16h*)(sB + cc * VQ_D + hi * 16);
            const v16h b1 = *(const v16h*)(sB + cc * VQ_D + 32 + hi * 16);
            v8f z = {};
            v8f d = __builtin_amdgcn_wmma_f32_16x16x32_f16(false, a0, false, b0,
                                                           (short)0, z, false, false);
            return __builtin_amdgcn_wmma_f32_16x16x32_f16(false, a1, false, b1,
                                                          (short)0, d, false, false);
        };
        // Argmin update for a tile's distances.
        auto upd = [&](const v8f& d, int t) {
            const float e  = en2reg[t];
            const int   cc = t * 16 + col;
#pragma unroll
            for (int r = 0; r < 8; ++r) {
                const float nd = d[r] + e;
                const bool lt = nd < minv[r];
                minv[r] = lt ? nd : minv[r];
                mini[r] = lt ? cc : mini[r];
            }
        };

        // ---- 2-stage modulo-scheduled loop over 32 column tiles ----
        v8f d0 = dist_tile(0);
        v8f d1;
#pragma unroll 1
        for (int tt = 0; tt < 15; ++tt) {
            const int base = 2 * tt;
            d1 = dist_tile(base + 1);
            upd(d0, base);          // overlaps d1's WMMAs
            d0 = dist_tile(base + 2);
            upd(d1, base + 1);      // overlaps d0's WMMAs
        }
        d1 = dist_tile(31);
        upd(d0, 30);
        upd(d1, 31);

        // Sortable keys: ordered-uint(dist) in high word, code in low word.
        unsigned long long key[8];
#pragma unroll
        for (int r = 0; r < 8; ++r) {
            const int bits = __float_as_int(minv[r]);
            const unsigned ou = (unsigned)bits ^ (unsigned)((bits >> 31) | 0x80000000);
            key[r] = ((unsigned long long)ou << 32) | (unsigned)mini[r];
        }

        // Branchless reduce across the 16 code-lanes of each half-wave.
#pragma unroll
        for (int m = 1; m < 16; m <<= 1) {
#pragma unroll
            for (int r = 0; r < 8; ++r) {
                const unsigned long long ok = __shfl_xor(key[r], m, 16);
                key[r] = (ok < key[r]) ? ok : key[r];
            }
        }

        // Indices: lane 0 -> rows 0..7, lane 16 -> rows 8..15.
        if (col == 0) {
            const int rbase = r0 + hi * 8;
#pragma unroll
            for (int r = 0; r < 8; ++r) idxOut[rbase + r] = (int)(unsigned)key[r];
        }

        // Gather codebook rows, write quantized, accumulate loss partial.
#pragma unroll
        for (int row = 0; row < 16; ++row) {
            const int idx = __shfl((int)(unsigned)key[row & 7], (row < 8) ? 0 : 16, 32);
            const float* q  = cb  + (size_t)idx * VQ_D + lane * 2;
            const float* xr = x   + (size_t)(r0 + row) * VQ_D + lane * 2;
            float*       o  = out + (size_t)(r0 + row) * VQ_D + lane * 2;
            const float q0 = q[0],  q1 = q[1];
            const float x0 = xr[0], x1 = xr[1];
            o[0] = q0; o[1] = q1;
            const float d0s = q0 - x0, d1s = q1 - x1;
            lossAcc = fmaf(d0s, d0s, lossAcc);
            lossAcc = fmaf(d1s, d1s, lossAcc);
        }
    }

    // Wave-level loss reduction; one partial per wave (deterministic).
#pragma unroll
    for (int m = 1; m < 32; m <<= 1) lossAcc += __shfl_xor(lossAcc, m, 32);
    if (lane == 0) partials[gwave] = lossAcc;
}

// ---------------------------------------------------------------------------
// Kernel 2: deterministic final loss reduction.
// loss = (1 + COMMIT) * sum((q-x)^2) / (N*D)
// ---------------------------------------------------------------------------
__global__ __launch_bounds__(256) void vq_loss_reduce_kernel(
    const float* __restrict__ partials, int n, float* __restrict__ lossOut) {
    __shared__ float s[256];
    float acc = 0.0f;
    for (int i = threadIdx.x; i < n; i += 256) acc += partials[i];
    s[threadIdx.x] = acc;
    __syncthreads();
    for (int off = 128; off > 0; off >>= 1) {
        if (threadIdx.x < off) s[threadIdx.x] += s[threadIdx.x + off];
        __syncthreads();
    }
    if (threadIdx.x == 0)
        *lossOut = s[0] * ((1.0f + COMMIT) / (float)((size_t)VQ_N * VQ_D));
}

// ---------------------------------------------------------------------------
extern "C" void kernel_launch(void* const* d_in, const int* in_sizes, int n_in,
                              void* d_out, int out_size, void* d_ws, size_t ws_size,
                              hipStream_t stream) {
    const float* x  = (const float*)d_in[0];   // [N, 64] f32
    const float* cb = (const float*)d_in[1];   // [512, 64] f32

    // d_out: quantized [N*64] f32, loss [1] f32, indices [N] i32 (raw bits)
    float* out     = (float*)d_out;
    float* lossOut = out + (size_t)VQ_N * VQ_D;
    int*   idxOut  = (int*)(lossOut + 1);

    // d_ws: [0..511] ||e||^2 (padded to 1024 floats), then per-wave partials
    float* en2      = (float*)d_ws;
    float* partials = en2 + 1024;

    vq_en2_kernel<<<2, 256, 0, stream>>>(cb, en2);
    vq_main_kernel<<<VQ_BLOCKS, VQ_TPB, 0, stream>>>(x, cb, en2, out, idxOut, partials);
    vq_loss_reduce_kernel<<<1, 256, 0, stream>>>(partials, VQ_WAVES, lossOut);
}
